// PCN_38336878084192
// MI455X (gfx1250) — matrix operand order repaired and checked by reading, not compiled, not run on hardware
//
#include <hip/hip_runtime.h>
#include <hip/hip_bf16.h>

// ---------------------------------------------------------------------------
// PCN: 3-layer MLP with synthesized weights, fused bf16-WMMA implementation
// for gfx1250 (MI455X).
//
// Layer sizes: 256 -> 512 -> 512 -> 256, batch 131072.
// ~137 GFLOP total, AI ~512 flop/byte when fused -> compute-bound, so use
// v_wmma_f32_16x16x32_bf16 (f32 accumulate), keep activations in LDS, and
// register-block 2x2 output tiles per wave to halve LDS/L2 fragment traffic
// per WMMA (4 wmma per 4 b128 fragment loads).
// ---------------------------------------------------------------------------

typedef __attribute__((ext_vector_type(16))) __bf16 v16bf;
typedef __attribute__((ext_vector_type(8)))  float  v8f;

#define BATCH   131072
#define M_TILE  64          // batch rows per workgroup
#define NWAVES  8           // 256 threads
#define L0      256
#define L1      512
#define L2W     512
#define L3      256

// ---------------------------------------------------------------------------
// Weight synthesis: Wt[n*K + k] = tri(||p_i[k] - p_{i+1}[n]||) / sqrt(K), bf16
// (column-major so WMMA B fragments are contiguous 16B loads per lane).
// tri(x) = 10 * (0.05 - |fmod(x, 0.2) - 0.1|)  for x >= 0  (PERIOD=0.1, AMP=1)
// ---------------------------------------------------------------------------
__device__ __forceinline__ float synth_elem(const float* __restrict__ pk,
                                            const float* __restrict__ pn,
                                            int k, int n, float scale) {
    float d2 = 0.0f;
#pragma unroll
    for (int d = 0; d < 20; ++d) {
        float t = pk[k * 20 + d] - pn[n * 20 + d];
        d2 += t * t;
    }
    float dist = sqrtf(fmaxf(d2, 0.0f));
    float m    = fmodf(dist, 0.2f);
    float tv   = 10.0f * (0.05f - fabsf(m - 0.1f));
    return tv * scale;
}

__global__ __launch_bounds__(256)
void pcn_synth_weights(const float* __restrict__ p0, const float* __restrict__ p1,
                       const float* __restrict__ p2, const float* __restrict__ p3,
                       __bf16* __restrict__ w0, __bf16* __restrict__ w1,
                       __bf16* __restrict__ w2) {
    int idx = blockIdx.x * blockDim.x + threadIdx.x;
    const int S0 = L1 * L0;   // 512*256 = 131072
    const int S1 = L2W * L1;  // 512*512 = 262144
    const int S2 = L3 * L2W;  // 256*512 = 131072
    if (idx < S0) {
        int n = idx / L0, k = idx % L0;                       // K=256
        w0[n * L0 + k] = (__bf16)synth_elem(p0, p1, k, n, 0.0625f);           // 1/sqrt(256)
    } else if (idx < S0 + S1) {
        int r = idx - S0;
        int n = r / L1, k = r % L1;                           // K=512
        w1[n * L1 + k] = (__bf16)synth_elem(p1, p2, k, n, 0.044194173824159216f); // 1/sqrt(512)
    } else if (idx < S0 + S1 + S2) {
        int r = idx - S0 - S1;
        int n = r / L2W, k = r % L2W;                         // K=512
        w2[n * L2W + k] = (__bf16)synth_elem(p2, p3, k, n, 0.044194173824159216f);
    }
}

// ---------------------------------------------------------------------------
// WMMA fragment loaders (wave32 layouts, CDNA5 ISA §7.12.2).
// ---------------------------------------------------------------------------
// A 16x32 bf16: lane r = row M (both halves); half h: chunks at k = 8h..8h+7
// and k = 16+8h..23+8h  -> two 16B loads.
__device__ __forceinline__ v16bf load_a_frag(const __bf16* row_k0, int half) {
    union { v16bf v; uint4 u[2]; } f;
    f.u[0] = *(const uint4*)(row_k0 + (half ? 8 : 0));
    f.u[1] = *(const uint4*)(row_k0 + 16 + (half ? 8 : 0));
    return f.v;
}
// B 32x16 bf16: lane r = col N; half h: chunks at k = 8h..8h+7
// and k = 16+8h..23+8h -> two 16B loads (weights stored column-major).
__device__ __forceinline__ v16bf load_b_frag(const __bf16* col_k0, int half) {
    union { v16bf v; uint4 u[2]; } f;
    f.u[0] = *(const uint4*)(col_k0 + 8 * half);
    f.u[1] = *(const uint4*)(col_k0 + 16 + 8 * half);
    return f.v;
}

#define WMMA_BF16(A, B, C) \
    __builtin_amdgcn_wmma_f32_16x16x32_bf16(false, (A), false, (B), (short)0, (C), false, false)

// ---------------------------------------------------------------------------
// One fused layer: act_in [64 x K] bf16 (LDS) x Wt [N x K] bf16 (global,
// column-major) -> 2x2-tile register-blocked WMMA, bias + optional relu fused.
// Output either to LDS as bf16 (OUT_F32=false) or to global f32 (OUT_F32=true).
// Block count (2 x N/32) is divisible by 8 waves -> uniform EXEC at each WMMA.
// ---------------------------------------------------------------------------
template<int K, int N, bool RELU, bool OUT_F32>
__device__ __forceinline__
void layer_compute(const __bf16* __restrict__ act_in,
                   __bf16* __restrict__ act_out_lds,
                   float* __restrict__ act_out_glb,
                   const __bf16* __restrict__ wt,
                   const float* __restrict__ bias) {
    const int lane = threadIdx.x & 31;
    const int wave = threadIdx.x >> 5;
    const int half = lane >> 4;
    const int r    = lane & 15;
    constexpr int MBLK = (M_TILE / 16) / 2;   // 2 blocks of 2 M-tiles
    constexpr int NBLK = (N / 16) / 2;        // 16 (N=512) or 8 (N=256)
    for (int t = wave; t < MBLK * NBLK; t += NWAVES) {
        const int mb  = t % MBLK;
        const int nb  = t / MBLK;
        const int mr0 = mb * 32;              // block row base
        const int nc0 = nb * 32;              // block col base
        const __bf16* arow0 = act_in + (mr0 + r) * K;
        const __bf16* arow1 = act_in + (mr0 + 16 + r) * K;
        const __bf16* bcol0 = wt + (size_t)(nc0 + r) * K;
        const __bf16* bcol1 = wt + (size_t)(nc0 + 16 + r) * K;
        v8f acc00 = {}, acc01 = {}, acc10 = {}, acc11 = {};
#pragma unroll
        for (int kb = 0; kb < K / 32; ++kb) {
            v16bf a0 = load_a_frag(arow0 + kb * 32, half);
            v16bf a1 = load_a_frag(arow1 + kb * 32, half);
            v16bf b0 = load_b_frag(bcol0 + kb * 32, half);
            v16bf b1 = load_b_frag(bcol1 + kb * 32, half);
            acc00 = WMMA_BF16(a0, b0, acc00);
            acc01 = WMMA_BF16(a0, b1, acc01);
            acc10 = WMMA_BF16(a1, b0, acc10);
            acc11 = WMMA_BF16(a1, b1, acc11);
        }
        const float bv0 = bias[nc0 + r];
        const float bv1 = bias[nc0 + 16 + r];
#pragma unroll
        for (int v = 0; v < 8; ++v) {
            const int row0 = mr0 + v + 8 * half;        // tiles (0,*)
            const int row1 = mr0 + 16 + v + 8 * half;   // tiles (1,*)
            float x00 = acc00[v] + bv0;
            float x01 = acc01[v] + bv1;
            float x10 = acc10[v] + bv0;
            float x11 = acc11[v] + bv1;
            if (RELU) {
                x00 = fmaxf(x00, 0.0f); x01 = fmaxf(x01, 0.0f);
                x10 = fmaxf(x10, 0.0f); x11 = fmaxf(x11, 0.0f);
            }
            if (OUT_F32) {
                act_out_glb[(size_t)row0 * N + nc0 + r]      = x00;
                act_out_glb[(size_t)row0 * N + nc0 + 16 + r] = x01;
                act_out_glb[(size_t)row1 * N + nc0 + r]      = x10;
                act_out_glb[(size_t)row1 * N + nc0 + 16 + r] = x11;
            } else {
                act_out_lds[row0 * N + nc0 + r]      = (__bf16)x00;
                act_out_lds[row0 * N + nc0 + 16 + r] = (__bf16)x01;
                act_out_lds[row1 * N + nc0 + r]      = (__bf16)x10;
                act_out_lds[row1 * N + nc0 + 16 + r] = (__bf16)x11;
            }
        }
    }
}

// ---------------------------------------------------------------------------
// Fused 3-layer MLP. Dynamic LDS: two 64x512 bf16 activation buffers (128 KB).
// ---------------------------------------------------------------------------
__global__ __launch_bounds__(256)
void pcn_mlp(const float* __restrict__ x,
             const __bf16* __restrict__ w0, const __bf16* __restrict__ w1,
             const __bf16* __restrict__ w2,
             const float* __restrict__ b1, const float* __restrict__ b2,
             const float* __restrict__ b3,
             float* __restrict__ out) {
    extern __shared__ __bf16 smem[];
    __bf16* bufA = smem;                     // up to 64 x 512
    __bf16* bufB = smem + M_TILE * 512;      // up to 64 x 512

    const int m0 = blockIdx.x * M_TILE;

    // Stage x tile (64 x 256 f32) -> bufA as bf16, float4-vectorized.
    for (int i = threadIdx.x; i < M_TILE * L0 / 4; i += blockDim.x) {
        const int e   = i * 4;
        const int row = e / L0;
        const int col = e % L0;
        float4 f = *(const float4*)(x + (size_t)(m0 + row) * L0 + col);
        __bf16* d = bufA + row * L0 + col;
        d[0] = (__bf16)f.x; d[1] = (__bf16)f.y;
        d[2] = (__bf16)f.z; d[3] = (__bf16)f.w;
    }
    __syncthreads();

    layer_compute<L0, L1, true, false>(bufA, bufB, nullptr, w0, b1);   // 256->512, relu
    __syncthreads();
    layer_compute<L1, L2W, true, false>(bufB, bufA, nullptr, w1, b2);  // 512->512, relu
    __syncthreads();
    layer_compute<L2W, L3, false, true>(bufA, nullptr,                 // 512->256 -> HBM
                                        out + (size_t)m0 * L3, w2, b3);
}

// ---------------------------------------------------------------------------
// Host-side launch. Input order: x, p0, p1, p2, p3, b1, b2, b3.
// Workspace holds the three synthesized bf16 weight matrices (~1 MB).
// ---------------------------------------------------------------------------
extern "C" void kernel_launch(void* const* d_in, const int* in_sizes, int n_in,
                              void* d_out, int out_size, void* d_ws, size_t ws_size,
                              hipStream_t stream) {
    (void)in_sizes; (void)n_in; (void)out_size; (void)ws_size;
    const float* x  = (const float*)d_in[0];
    const float* p0 = (const float*)d_in[1];
    const float* p1 = (const float*)d_in[2];
    const float* p2 = (const float*)d_in[3];
    const float* p3 = (const float*)d_in[4];
    const float* b1 = (const float*)d_in[5];
    const float* b2 = (const float*)d_in[6];
    const float* b3 = (const float*)d_in[7];

    char* ws = (char*)d_ws;
    __bf16* w0 = (__bf16*)ws;                                   // [512 x 256]
    __bf16* w1 = (__bf16*)(ws + (size_t)L1 * L0 * 2);           // [512 x 512]
    __bf16* w2 = (__bf16*)(ws + (size_t)L1 * L0 * 2 + (size_t)L2W * L1 * 2); // [256 x 512]

    // 1) synthesize weights (524288 elements)
    const int total = L1 * L0 + L2W * L1 + L3 * L2W;
    pcn_synth_weights<<<(total + 255) / 256, 256, 0, stream>>>(p0, p1, p2, p3,
                                                               w0, w1, w2);

    // 2) fused 3-layer WMMA MLP; 128 KB dynamic LDS per workgroup
    const size_t lds_bytes = (size_t)2 * M_TILE * 512 * sizeof(__bf16);
    pcn_mlp<<<BATCH / M_TILE, 256, lds_bytes, stream>>>(x, w0, w1, w2,
                                                        b1, b2, b3,
                                                        (float*)d_out);
}
